// QCMAEDecoder_45861660787328
// MI455X (gfx1250) — matrix operand order repaired and checked by reading, not compiled, not run on hardware
//
#include <hip/hip_runtime.h>
#include <hip/hip_bf16.h>

// ---------------------------------------------------------------------------
// QCMAE decoder forward for gfx1250 (MI455X, wave32, WMMA).
// Dense GEMMs -> v_wmma_f32_16x16x32_f16. Weights (B) are pre-packed to f16
// in fragment-major layout: one contiguous 32-byte record per (kblock, ntile,
// lane-half, column) -> a single v16h load per tile per k-step in the hot
// loop. Activations (A) are streamed f32 -> cvt_pk_f16. K % 32 == 0
// everywhere (fourier "x" row is a rank-1 epilogue), M % 32 == 0.
// ---------------------------------------------------------------------------

typedef __attribute__((ext_vector_type(16))) _Float16 v16h;
typedef __attribute__((ext_vector_type(8)))  float    v8f;

namespace {

constexpr int A_    = 512;
constexpr int P_    = 2048;
constexpr int T_    = 50;
constexpr int NT2M_ = 30;
constexpr int FUT_  = 60;
constexpr int KM    = 6;
constexpr int RR    = 3;
constexpr int LL    = 2;
constexpr int H     = 128;
constexpr int NFB_  = 64;
constexpr int OD_   = 2;
constexpr int DEGPL = 32;
constexpr int DEGA  = 16;
constexpr int AKN   = A_ * KM;          // 3072 mode nodes
constexpr float PI_F  = 3.14159265358979323846f;
constexpr float TPI_F = 6.28318530717958647692f;

enum { GF_BIAS = 1, GF_RELU = 2, GF_ACC = 4, GF_R1 = 8 };
enum { LN_PLAIN = 0, LN_RELU = 1, LN_ADD = 2 };

// --------------------------- B packing (f32 -> f16) ------------------------
// Record layout: slot = ((kb*nT2 + nb)*2 + half)*16 + col, each slot = 16 f16
// = K rows kb*32 + half*8 + j (j < 8) and kb*32 + 16 + half*8 + (j - 8) (j >= 8) of column (nb*16 + col).
// Pad columns (>= N) are zero. nT2 is even so a wave's 2nd tile is in-bounds.
__global__ void pack_b(const float* __restrict__ B, int ldb, int N, int Kd, int nT2,
                       _Float16* __restrict__ Bp) {
  int idx = blockIdx.x * 256 + threadIdx.x;
  int total = (Kd >> 5) * nT2 * 32;            // records
  if (idx >= total) return;
  int r    = idx & 15;
  int half = (idx >> 4) & 1;
  int rest = idx >> 5;
  int nb   = rest % nT2;
  int kb   = rest / nT2;
  int col  = nb * 16 + r;
  int krow = kb * 32 + half * 8;
  _Float16* out = Bp + (size_t)idx * 16;
  #pragma unroll
  for (int j = 0; j < 16; ++j) {
    float v = (col < N) ? B[(size_t)(krow + j + (j >= 8 ? 8 : 0)) * ldb + col] : 0.f;
    out[j] = (_Float16)v;
  }
}

// ------------------------------ GEMM (WMMA) --------------------------------
// C[M,N] = act(A[M,K] @ B[K,N] + bias + r1a[row]*r1b[col] (+C))
// B is pre-packed (see pack_b). Each wave computes 16x32 (one A fragment
// feeds two WMMAs); a block of 8 waves tiles 32(M) x 128(N).
__device__ __forceinline__ void wmma_kstep2(
    const float* __restrict__ Ap, const v16h* __restrict__ b0,
    const v16h* __restrict__ b1, int kb, int kstr, int half,
    v8f& acc0, v8f& acc1)
{
  v16h af;
  int k0 = kb * 32;
  #pragma unroll
  for (int v = 0; v < 8; ++v) {
    int k = k0 + ((v < 4) ? (2 * v + 8 * half) : (16 + 2 * (v - 4) + 8 * half));
    float2 x2 = *reinterpret_cast<const float2*>(Ap + k);
    af[2 * v]     = (_Float16)x2.x;
    af[2 * v + 1] = (_Float16)x2.y;
  }
  v16h bf0 = b0[(size_t)kb * kstr];
  v16h bf1 = b1[(size_t)kb * kstr];
  acc0 = __builtin_amdgcn_wmma_f32_16x16x32_f16(false, af, false, bf0,
                                                (short)0, acc0, false, false);
  acc1 = __builtin_amdgcn_wmma_f32_16x16x32_f16(false, af, false, bf1,
                                                (short)0, acc1, false, false);
}

__global__ __launch_bounds__(256) void gemm_wmma(
    const float* __restrict__ Am, int lda,
    const _Float16* __restrict__ Bp, int nT2,
    const float* __restrict__ bias,
    const float* __restrict__ r1a, int r1as,
    const float* __restrict__ r1b,
    float* __restrict__ Cm, int ldc,
    int M, int N, int Kd, int flags)
{
  const int lane = threadIdx.x & 31;
  const int wv   = threadIdx.x >> 5;
  const int tm   = blockIdx.y * 32 + (wv >> 2) * 16;
  const int tn   = blockIdx.x * 128 + (wv & 3) * 32;
  if (tn >= N) return;                          // wave-uniform exit (no barriers)

  const int half  = lane >> 4;
  const int r     = lane & 15;
  const int arow  = tm + r;                     // always < M (M % 32 == 0)
  const int bcol0 = tn + r;
  const int bcol1 = tn + 16 + r;
  const int nb0   = tn >> 4;
  const float* __restrict__ Ap = Am + (size_t)arow * lda;
  const v16h* __restrict__ Bv = reinterpret_cast<const v16h*>(Bp);
  const v16h* __restrict__ b0 = Bv + ((size_t)nb0 * 2 + half) * 16 + r;
  const v16h* __restrict__ b1 = Bv + ((size_t)(nb0 + 1) * 2 + half) * 16 + r;
  const int kstr = nT2 * 32;

  v8f acc0 = {}, acc1 = {};
  if (Kd == 128) {
    #pragma unroll
    for (int kb = 0; kb < 4; ++kb)
      wmma_kstep2(Ap, b0, b1, kb, kstr, half, acc0, acc1);
  } else {
    int nkb = Kd >> 5;
    for (int kb = 0; kb < nkb; ++kb)
      wmma_kstep2(Ap, b0, b1, kb, kstr, half, acc0, acc1);
  }

  #pragma unroll
  for (int tile = 0; tile < 2; ++tile) {
    int bcol = tile ? bcol1 : bcol0;
    if (bcol >= N) continue;
    v8f& acc = tile ? acc1 : acc0;
    float bv = (flags & GF_BIAS) ? bias[bcol] : 0.f;
    float rb = (flags & GF_R1)   ? r1b[bcol]  : 0.f;
    #pragma unroll
    for (int j = 0; j < 8; ++j) {
      int row = tm + j + 8 * half;
      float v = acc[j] + bv;
      if (flags & GF_R1)   v += r1a[(size_t)row * r1as] * rb;
      if (flags & GF_ACC)  v += Cm[(size_t)row * ldc + bcol];
      if (flags & GF_RELU) v = v > 0.f ? v : 0.f;
      Cm[(size_t)row * ldc + bcol] = v;
    }
  }
}

// ------------------------------ LayerNorm (H=128) --------------------------
__global__ __launch_bounds__(256) void layernorm128(
    const float* __restrict__ X, int ldx,
    const float* __restrict__ g, const float* __restrict__ b,
    const float* __restrict__ res, int ldr,
    float* __restrict__ O, int ldo, int Nrows, int mode)
{
  int row = blockIdx.x * 8 + (threadIdx.x >> 5);
  if (row >= Nrows) return;
  int lane = threadIdx.x & 31;
  const float* x = X + (size_t)row * ldx;
  float v[4], s = 0.f;
  #pragma unroll
  for (int j = 0; j < 4; ++j) { v[j] = x[lane + 32 * j]; s += v[j]; }
  #pragma unroll
  for (int o = 16; o >= 1; o >>= 1) s += __shfl_xor(s, o, 32);
  float mu = s * (1.f / 128.f);
  float q = 0.f;
  #pragma unroll
  for (int j = 0; j < 4; ++j) { float d = v[j] - mu; q += d * d; }
  #pragma unroll
  for (int o = 16; o >= 1; o >>= 1) q += __shfl_xor(q, o, 32);
  float rstd = rsqrtf(q * (1.f / 128.f) + 1e-5f);
  float* oo = O + (size_t)row * ldo;
  #pragma unroll
  for (int j = 0; j < 4; ++j) {
    int c = lane + 32 * j;
    float y = (v[j] - mu) * rstd * g[c] + b[c];
    if (mode == LN_RELU)      y = y > 0.f ? y : 0.f;
    else if (mode == LN_ADD)  y += res[(size_t)row * ldr + c];
    oo[c] = y;
  }
}

// --------------------- fixed-degree multi-head attention -------------------
__global__ __launch_bounds__(256) void attn_agg(
    const float* __restrict__ Q,
    const float* __restrict__ Kc, const float* __restrict__ Vc,
    const float* __restrict__ KR, const float* __restrict__ VR,
    const int* __restrict__ esrc, const int* __restrict__ eridx,
    int deg, int Ndst, float* __restrict__ agg)
{
  int dst = blockIdx.x * 8 + (threadIdx.x >> 5);
  if (dst >= Ndst) return;
  int lane = threadIdx.x & 31;
  float q[4];
  #pragma unroll
  for (int j = 0; j < 4; ++j) q[j] = Q[(size_t)dst * 128 + lane + 32 * j];
  float mrun[4], den[4], acc[4];
  #pragma unroll
  for (int j = 0; j < 4; ++j) { mrun[j] = -1e30f; den[j] = 0.f; acc[j] = 0.f; }
  const int* ep = esrc + (size_t)dst * deg;
  const int* rp = eridx ? (eridx + (size_t)dst * deg) : nullptr;
  for (int e = 0; e < deg; ++e) {
    int s  = ep[e];
    int ri = rp ? rp[e] : 0;
    float kv[4], vv[4];
    #pragma unroll
    for (int j = 0; j < 4; ++j) {
      int c = lane + 32 * j;
      float kk = Kc[(size_t)s * 128 + c];
      float vx = Vc[(size_t)s * 128 + c];
      if (KR) { kk += KR[(size_t)ri * 128 + c]; vx += VR[(size_t)ri * 128 + c]; }
      kv[j] = kk; vv[j] = vx;
    }
    #pragma unroll
    for (int j = 0; j < 4; ++j) {
      float p = q[j] * kv[j];
      p += __shfl_xor(p, 1, 32);
      p += __shfl_xor(p, 2, 32);
      p += __shfl_xor(p, 4, 32);
      p += __shfl_xor(p, 8, 32);          // sum over the 16-lane head group
      float sim = p * 0.25f;              // DH^-0.5
      float mn  = fmaxf(mrun[j], sim);
      float rs  = __expf(mrun[j] - mn);
      float ee  = __expf(sim - mn);
      den[j] = den[j] * rs + ee;
      acc[j] = acc[j] * rs + ee * vv[j];
      mrun[j] = mn;
    }
  }
  #pragma unroll
  for (int j = 0; j < 4; ++j)
    agg[(size_t)dst * 128 + lane + 32 * j] = acc[j] / den[j];
}

// ------------------------------ small kernels ------------------------------
__device__ __forceinline__ float wrapang(float a) {
  float w = fmodf(a + PI_F, TPI_F);
  if (w < 0.f) w += TPI_F;
  return w - PI_F;
}

__global__ void feat_t2m(const float* __restrict__ pos_a,
                         const float* __restrict__ head_a,
                         float* __restrict__ f) {
  int i = blockIdx.x * 256 + threadIdx.x;
  if (i >= A_ * NT2M_) return;
  int a = i / NT2M_, j = i % NT2M_, t = T_ - NT2M_ + j;
  float px = pos_a[((size_t)a * T_ + t) * 2], py = pos_a[((size_t)a * T_ + t) * 2 + 1];
  float mx = pos_a[((size_t)a * T_ + T_ - 1) * 2], my = pos_a[((size_t)a * T_ + T_ - 1) * 2 + 1];
  float hm = head_a[(size_t)a * T_ + T_ - 1];
  float hx = cosf(hm), hy = sinf(hm);
  float rx = px - mx, ry = py - my;
  f[i * 4 + 0] = sqrtf(rx * rx + ry * ry);
  f[i * 4 + 1] = atan2f(hx * ry - hy * rx, hx * rx + hy * ry);
  f[i * 4 + 2] = wrapang(head_a[(size_t)a * T_ + t] - hm);
  f[i * 4 + 3] = (float)(t - (T_ - 1));
}

__global__ void feat_pl2m(const float* __restrict__ pos_a, const float* __restrict__ head_a,
                          const float* __restrict__ pos_pl, const float* __restrict__ orient_pl,
                          const int* __restrict__ srcmap, float* __restrict__ f) {
  int i = blockIdx.x * 256 + threadIdx.x;
  if (i >= A_ * DEGPL) return;
  int a = i / DEGPL, p = srcmap[i];
  float mx = pos_a[((size_t)a * T_ + T_ - 1) * 2], my = pos_a[((size_t)a * T_ + T_ - 1) * 2 + 1];
  float hm = head_a[(size_t)a * T_ + T_ - 1];
  float hx = cosf(hm), hy = sinf(hm);
  float rx = pos_pl[(size_t)p * 2] - mx, ry = pos_pl[(size_t)p * 2 + 1] - my;
  f[i * 3 + 0] = sqrtf(rx * rx + ry * ry);
  f[i * 3 + 1] = atan2f(hx * ry - hy * rx, hx * rx + hy * ry);
  f[i * 3 + 2] = wrapang(orient_pl[p] - hm);
}

__global__ void feat_a2m(const float* __restrict__ pos_a, const float* __restrict__ head_a,
                         const int* __restrict__ srcmap, float* __restrict__ f) {
  int i = blockIdx.x * 256 + threadIdx.x;
  if (i >= A_ * DEGA) return;
  int a = i / DEGA, s = srcmap[i];
  float mx = pos_a[((size_t)a * T_ + T_ - 1) * 2], my = pos_a[((size_t)a * T_ + T_ - 1) * 2 + 1];
  float hm = head_a[(size_t)a * T_ + T_ - 1];
  float sx = pos_a[((size_t)s * T_ + T_ - 1) * 2], sy = pos_a[((size_t)s * T_ + T_ - 1) * 2 + 1];
  float hs = head_a[(size_t)s * T_ + T_ - 1];
  float hx = cosf(hm), hy = sinf(hm);
  float rx = sx - mx, ry = sy - my;
  f[i * 3 + 0] = sqrtf(rx * rx + ry * ry);
  f[i * 3 + 1] = atan2f(hx * ry - hy * rx, hx * rx + hy * ry);
  f[i * 3 + 2] = wrapang(hs - hm);
}

__global__ void build_t2m_edges(int* __restrict__ es, int* __restrict__ er) {
  int i = blockIdx.x * 256 + threadIdx.x;
  if (i >= AKN * NT2M_) return;
  int j = i % NT2M_, d = i / NT2M_, a = d / KM;
  es[i] = a * T_ + (T_ - NT2M_ + j);
  er[i] = a * NT2M_ + j;
}

__global__ void build_deg_edges(const int* __restrict__ srcmap, int degd,
                                int* __restrict__ es, int* __restrict__ er) {
  int i = blockIdx.x * 256 + threadIdx.x;
  if (i >= AKN * degd) return;
  int j = i % degd, d = i / degd, a = d / KM;
  es[i] = srcmap[a * degd + j];
  er[i] = a * degd + j;
}

__global__ void build_m2m_edges(int* __restrict__ es) {
  int i = blockIdx.x * 256 + threadIdx.x;
  if (i >= AKN * KM) return;
  int j = i % KM, d = i / KM, a = d / KM;
  es[i] = a * KM + j;
}

// z = [cos(2*pi*x*f), sin(2*pi*x*f)] laid out [din][N][128]; the "x" term is
// applied as a rank-1 epilogue in the w1 GEMM.
__global__ void fourier_z(const float* __restrict__ X, int ldx,
                          const float* __restrict__ freqs,
                          float* __restrict__ Z, int N, int din) {
  int i = blockIdx.x * 256 + threadIdx.x;
  int total = N * din * NFB_;
  if (i >= total) return;
  int f = i % NFB_;
  int nd = i / NFB_;
  int d = nd % din, n = nd / din;
  float x = X[(size_t)n * ldx + d];
  float ang = x * freqs[d * NFB_ + f] * TPI_F;
  float* zr = Z + ((size_t)d * N + n) * 128;
  zr[f]        = cosf(ang);
  zr[NFB_ + f] = sinf(ang);
}

__global__ void gate_fuse(const float* __restrict__ agg, const float* __restrict__ g,
                          const float* __restrict__ S, float* __restrict__ o, int n) {
  int i = blockIdx.x * 256 + threadIdx.x;
  if (i >= n) return;
  float gg = 1.f / (1.f + __expf(-g[i]));
  float a = agg[i];
  o[i] = a + gg * (S[i] - a);
}

__global__ void gru_gate(const float* __restrict__ GI, const float* __restrict__ GH,
                         float* __restrict__ h, int n) {
  int i = blockIdx.x * 256 + threadIdx.x;
  if (i >= n) return;
  int row = i >> 7, c = i & 127;
  const float* gi = GI + (size_t)row * 384;
  const float* gh = GH + (size_t)row * 384;
  float r  = 1.f / (1.f + __expf(-(gi[c] + gh[c])));
  float z  = 1.f / (1.f + __expf(-(gi[128 + c] + gh[128 + c])));
  float ng = tanhf(gi[256 + c] + r * gh[256 + c]);
  float hv = h[i];
  h[i] = (1.f - z) * ng + z * hv;
}

__global__ void init_mode(const float* __restrict__ mode_emb, float* __restrict__ m) {
  int i = blockIdx.x * 256 + threadIdx.x;
  if (i >= AKN * H) return;
  int row = i >> 7, c = i & 127;
  m[i] = mode_emb[(row % KM) * H + c];
}

__global__ void bcast_row(const float* __restrict__ h0, float* __restrict__ h, int n) {
  int i = blockIdx.x * 256 + threadIdx.x;
  if (i >= n) return;
  h[i] = h0[i & 127];
}

__global__ void cumsum_out(const float* __restrict__ locs, const float* __restrict__ scales,
                           float* __restrict__ out_loc, float* __restrict__ out_scale) {
  int i = blockIdx.x * 256 + threadIdx.x;
  if (i >= AKN * OD_) return;
  int od = i & 1, ak = i >> 1;
  const float* lr = locs + (size_t)ak * 120;
  const float* sr = scales + (size_t)ak * 120;
  float cl = 0.f, cs = 0.f;
  for (int t = 0; t < FUT_; ++t) {
    cl += lr[t * 2 + od];
    float x = sr[t * 2 + od];
    float e = x > 0.f ? x : (__expf(x) - 1.f);
    cs += e + 1.f;
    out_loc[((size_t)ak * FUT_ + t) * 2 + od]   = cl;
    out_scale[((size_t)ak * FUT_ + t) * 2 + od] = cs + 0.1f;
  }
}

__global__ void ref_out(const float* __restrict__ locr, const float* __restrict__ scaler,
                        const float* __restrict__ loc_prop,
                        float* __restrict__ out_locref, float* __restrict__ out_scaleref) {
  int i = blockIdx.x * 256 + threadIdx.x;
  if (i >= AKN * FUT_ * OD_) return;
  out_locref[i] = locr[i] + loc_prop[i];
  float x = scaler[i];
  out_scaleref[i] = (x > 0.f ? x : (__expf(x) - 1.f)) + 1.1f;
}

// ------------------------------ host helpers -------------------------------
struct LinP  { const float *w, *b; };
struct LnP   { const float *g, *b; };
struct FourP { const float *freqs, *w1, *b1, *ln1g, *ln1b, *w2, *b2, *lnog, *lnob, *wow, *wob; };
struct MlpP  { const float *w1, *b1, *lng, *lnb, *w2, *b2; };
struct GruP  { const float *wih, *whh, *bih, *bhh; };
struct AttnP {
  LnP ln_s, ln_d, ln_post, ln_ff, ln_ffpost;
  LinP wq, wk, wv, ws, wg, wo, ff1, ff2;
  LnP ln_r; LinP wkr, wvr; bool has_r;
};
struct Cursor {
  void* const* din; int i;
  const float* f() { return (const float*)din[i++]; }
};
static LnP  ln_p(Cursor& c)  { LnP p;  p.g = c.f(); p.b = c.f(); return p; }
static LinP lin_p(Cursor& c, bool bias) { LinP p; p.w = c.f(); p.b = bias ? c.f() : nullptr; return p; }
static FourP four_p(Cursor& c) {
  FourP p; p.freqs = c.f(); p.w1 = c.f(); p.b1 = c.f(); p.ln1g = c.f(); p.ln1b = c.f();
  p.w2 = c.f(); p.b2 = c.f(); p.lnog = c.f(); p.lnob = c.f(); p.wow = c.f(); p.wob = c.f();
  return p;
}
static AttnP attn_p(Cursor& c, bool has_r) {
  AttnP p;
  p.ln_s = ln_p(c); p.ln_d = ln_p(c); p.ln_post = ln_p(c); p.ln_ff = ln_p(c); p.ln_ffpost = ln_p(c);
  p.wq = lin_p(c, true); p.wk = lin_p(c, false); p.wv = lin_p(c, true);
  p.ws = lin_p(c, true); p.wg = lin_p(c, true);  p.wo = lin_p(c, true);
  p.ff1 = lin_p(c, true); p.ff2 = lin_p(c, true);
  p.has_r = has_r;
  if (has_r) { p.ln_r = ln_p(c); p.wkr = lin_p(c, false); p.wvr = lin_p(c, true); }
  return p;
}
static MlpP mlp_p(Cursor& c) {
  MlpP p; p.w1 = c.f(); p.b1 = c.f(); p.lng = c.f(); p.lnb = c.f(); p.w2 = c.f(); p.b2 = c.f();
  return p;
}

static inline dim3 ew(int n) { return dim3((n + 255) / 256); }
static inline int  ntiles2(int N) { return ((N + 31) / 32) * 2; }  // even #16-col tiles

static inline void pack_weights(hipStream_t s, const float* B, int ldb, int N, int Kd,
                                _Float16* Bp) {
  int records = (Kd >> 5) * ntiles2(N) * 32;
  pack_b<<<ew(records), 256, 0, s>>>(B, ldb, N, Kd, ntiles2(N), Bp);
}
static inline void gemm_packed(hipStream_t s, const float* A, int lda,
                               const _Float16* Bp, const float* bias,
                               const float* r1a, int r1as, const float* r1b,
                               float* C, int ldc, int M, int N, int Kd, int flags) {
  dim3 g((N + 127) / 128, (M + 31) / 32);
  gemm_wmma<<<g, 256, 0, s>>>(A, lda, Bp, ntiles2(N), bias, r1a, r1as, r1b,
                              C, ldc, M, N, Kd, flags);
}
static inline void gemm_r1(hipStream_t s, const float* A, int lda, const float* B, int ldb,
                           const float* bias, const float* r1a, int r1as, const float* r1b,
                           float* C, int ldc, int M, int N, int Kd, int flags,
                           _Float16* Bp) {
  pack_weights(s, B, ldb, N, Kd, Bp);
  gemm_packed(s, A, lda, Bp, bias, r1a, r1as, r1b, C, ldc, M, N, Kd, flags);
}
static inline void gemm(hipStream_t s, const float* A, int lda, const float* B, int ldb,
                        const float* bias, float* C, int ldc, int M, int N, int Kd,
                        int flags, _Float16* Bp) {
  gemm_r1(s, A, lda, B, ldb, bias, nullptr, 0, nullptr, C, ldc, M, N, Kd, flags, Bp);
}
static inline void ln128(hipStream_t s, const float* X, int ldx, const float* g, const float* b,
                         const float* res, int ldr, float* O, int ldo, int N, int mode) {
  layernorm128<<<dim3((N + 7) / 8), 256, 0, s>>>(X, ldx, g, b, res, ldr, O, ldo, N, mode);
}

struct Scratch {
  float *XS, *Kc, *V, *RN, *KR, *VR, *XD, *Q, *agg, *g, *S, *o, *O, *H1;
  _Float16 *Bp;   // packed-weight scratch (reused by every unpacked GEMM)
};

static void fourier_forward(hipStream_t st, const FourP& p, const float* X, int ldx,
                            int N, int din, float* Z, float* hbuf, float* sbuf,
                            float* out, _Float16* Bp) {
  fourier_z<<<ew(N * din * NFB_), 256, 0, st>>>(X, ldx, p.freqs, Z, N, din);
  for (int d = 0; d < din; ++d) {
    // h = z_cos_sin @ w1[d][0:128] + x * w1[d][128] + b1[d]   (rank-1 x-term)
    gemm_r1(st, Z + (size_t)d * N * H, H, p.w1 + (size_t)d * 129 * H, H,
            p.b1 + d * H, X + d, ldx, p.w1 + (size_t)d * 129 * H + 128 * H,
            hbuf, H, N, H, H, GF_BIAS | GF_R1, Bp);
    ln128(st, hbuf, H, p.ln1g + d * H, p.ln1b + d * H, nullptr, 0, hbuf, H, N, LN_RELU);
    gemm(st, hbuf, H, p.w2 + (size_t)d * H * H, H, p.b2 + d * H,
         sbuf, H, N, H, H, GF_BIAS | (d ? GF_ACC : 0), Bp);
  }
  ln128(st, sbuf, H, p.lnog, p.lnob, nullptr, 0, sbuf, H, N, LN_RELU);
  gemm(st, sbuf, H, p.wow, H, p.wob, out, H, N, H, H, GF_BIAS, Bp);
}

static void attn_apply(hipStream_t st, const AttnP& p,
                       const float* xsrc, int lds, int Nsrc,
                       float* m,
                       const float* remb, int Nr,
                       const int* esrc, const int* er, int deg,
                       bool bipartite, const Scratch& w) {
  _Float16* Bp = w.Bp;
  ln128(st, xsrc, lds, p.ln_s.g, p.ln_s.b, nullptr, 0, w.XS, H, Nsrc, LN_PLAIN);
  const float* XD;
  if (bipartite) {
    ln128(st, m, H, p.ln_d.g, p.ln_d.b, nullptr, 0, w.XD, H, AKN, LN_PLAIN);
    XD = w.XD;
  } else {
    XD = w.XS;
  }
  gemm(st, XD, H, p.wq.w, H, p.wq.b, w.Q, H, AKN, H, H, GF_BIAS, Bp);
  gemm(st, w.XS, H, p.wk.w, H, nullptr, w.Kc, H, Nsrc, H, H, 0, Bp);
  gemm(st, w.XS, H, p.wv.w, H, p.wv.b, w.V, H, Nsrc, H, H, GF_BIAS, Bp);
  const float *KRp = nullptr, *VRp = nullptr;
  if (p.has_r && remb) {
    ln128(st, remb, H, p.ln_r.g, p.ln_r.b, nullptr, 0, w.RN, H, Nr, LN_PLAIN);
    gemm(st, w.RN, H, p.wkr.w, H, nullptr, w.KR, H, Nr, H, H, 0, Bp);
    gemm(st, w.RN, H, p.wvr.w, H, p.wvr.b, w.VR, H, Nr, H, H, GF_BIAS, Bp);
    KRp = w.KR; VRp = w.VR;
  }
  attn_agg<<<dim3((AKN + 7) / 8), 256, 0, st>>>(w.Q, w.Kc, w.V, KRp, VRp,
                                                esrc, p.has_r ? er : nullptr,
                                                deg, AKN, w.agg);
  // g = sigmoid([agg | xd] @ wg + b) split into two GEMMs (wg is [256,128]).
  gemm(st, w.agg, H, p.wg.w, H, p.wg.b, w.g, H, AKN, H, H, GF_BIAS, Bp);
  gemm(st, XD, H, p.wg.w + 128 * H, H, nullptr, w.g, H, AKN, H, H, GF_ACC, Bp);
  gemm(st, XD, H, p.ws.w, H, p.ws.b, w.S, H, AKN, H, H, GF_BIAS, Bp);
  int n = AKN * H;
  gate_fuse<<<ew(n), 256, 0, st>>>(w.agg, w.g, w.S, w.o, n);
  gemm(st, w.o, H, p.wo.w, H, p.wo.b, w.O, H, AKN, H, H, GF_BIAS, Bp);
  ln128(st, w.O, H, p.ln_post.g, p.ln_post.b, m, H, m, H, AKN, LN_ADD);
  ln128(st, m, H, p.ln_ff.g, p.ln_ff.b, nullptr, 0, w.Q, H, AKN, LN_PLAIN);  // Q as Hf
  gemm(st, w.Q, H, p.ff1.w, 512, p.ff1.b, w.H1, 512, AKN, 512, H, GF_BIAS | GF_RELU, Bp);
  gemm(st, w.H1, 512, p.ff2.w, H, p.ff2.b, w.O, H, AKN, H, 512, GF_BIAS, Bp);
  ln128(st, w.O, H, p.ln_ffpost.g, p.ln_ffpost.b, m, H, m, H, AKN, LN_ADD);
}

static void mlp_head_run(hipStream_t st, const MlpP& p, const float* x, float* htmp,
                         float* out, int ldo, int dout, _Float16* Bp) {
  gemm(st, x, H, p.w1, H, p.b1, htmp, H, AKN, H, H, GF_BIAS, Bp);
  ln128(st, htmp, H, p.lng, p.lnb, nullptr, 0, htmp, H, AKN, LN_RELU);
  gemm(st, htmp, H, p.w2, dout, p.b2, out, ldo, AKN, dout, H, GF_BIAS, Bp);
}

} // namespace

// ------------------------------- entry point -------------------------------
extern "C" void kernel_launch(void* const* d_in, const int* in_sizes, int n_in,
                              void* d_out, int out_size, void* d_ws, size_t ws_size,
                              hipStream_t stream) {
  (void)in_sizes; (void)n_in; (void)out_size; (void)ws_size;
  hipStream_t st = stream;

  const float* pos_a     = (const float*)d_in[0];
  const float* head_a    = (const float*)d_in[1];
  const float* pos_pl    = (const float*)d_in[2];
  const float* orient_pl = (const float*)d_in[3];
  const float* x_a_enc   = (const float*)d_in[4];
  const float* x_pl      = (const float*)d_in[5];
  const int*   pl2m_src  = (const int*)d_in[6];
  const int*   a2m_src   = (const int*)d_in[7];

  Cursor c{d_in, 8};
  const float* mode_emb = c.f();
  FourP Pt2m = four_p(c);
  FourP Ppl  = four_p(c);
  FourP Pa2m = four_p(c);
  FourP Py   = four_p(c);
  const float* h0 = c.f();
  GruP gru; gru.wih = c.f(); gru.whh = c.f(); gru.bih = c.f(); gru.bhh = c.f();
  AttnP t2m_prop[LL]  = { attn_p(c, true), attn_p(c, true) };
  AttnP pl2m_prop[LL] = { attn_p(c, true), attn_p(c, true) };
  AttnP a2m_prop[LL]  = { attn_p(c, true), attn_p(c, true) };
  AttnP m2m_prop      = attn_p(c, false);
  AttnP t2m_ref[LL]   = { attn_p(c, true), attn_p(c, true) };
  AttnP pl2m_ref[LL]  = { attn_p(c, true), attn_p(c, true) };
  AttnP a2m_ref[LL]   = { attn_p(c, true), attn_p(c, true) };
  AttnP m2m_ref       = attn_p(c, false);
  MlpP to_loc_prop   = mlp_p(c);
  MlpP to_scale_prop = mlp_p(c);
  MlpP to_loc_ref    = mlp_p(c);
  MlpP to_scale_ref  = mlp_p(c);
  MlpP to_pi         = mlp_p(c);

  // ----- workspace bump allocator -----
  size_t off = 0;
  auto af = [&](size_t n) -> float* {
    float* p = (float*)((char*)d_ws + off);
    off += (n * sizeof(float) + 255) & ~(size_t)255;
    return p;
  };
  auto ai = [&](size_t n) -> int* {
    int* p = (int*)((char*)d_ws + off);
    off += (n * sizeof(int) + 255) & ~(size_t)255;
    return p;
  };
  auto ah = [&](size_t n) -> _Float16* {   // n f16 elements
    _Float16* p = (_Float16*)((char*)d_ws + off);
    off += (n * sizeof(_Float16) + 255) & ~(size_t)255;
    return p;
  };

  float* remb_t2m = af((size_t)A_ * NT2M_ * H);
  float* remb_pl  = af((size_t)A_ * DEGPL * H);
  float* remb_a   = af((size_t)A_ * DEGA * H);
  float* mbuf     = af((size_t)AKN * H);
  float* locs     = af((size_t)AKN * 120);
  float* scales   = af((size_t)AKN * 120);
  float* f_t2m    = af((size_t)A_ * NT2M_ * 4);
  float* f_pl     = af((size_t)A_ * DEGPL * 3);
  float* f_a      = af((size_t)A_ * DEGA * 3);
  int* es_t2m = ai((size_t)AKN * NT2M_);
  int* er_t2m = ai((size_t)AKN * NT2M_);
  int* es_pl  = ai((size_t)AKN * DEGPL);
  int* er_pl  = ai((size_t)AKN * DEGPL);
  int* es_a   = ai((size_t)AKN * DEGA);
  int* er_a   = ai((size_t)AKN * DEGA);
  int* es_m2m = ai((size_t)AKN * KM);
  float* Zb = af((size_t)4 * 16384 * H);
  float* hb = af((size_t)16384 * H);
  float* sb = af((size_t)16384 * H);
  Scratch w;
  w.XS  = af((size_t)A_ * T_ * H);
  w.Kc  = af((size_t)A_ * T_ * H);
  w.V   = af((size_t)A_ * T_ * H);
  w.RN  = af((size_t)A_ * DEGPL * H);
  w.KR  = af((size_t)A_ * DEGPL * H);
  w.VR  = af((size_t)A_ * DEGPL * H);
  w.XD  = af((size_t)AKN * H);
  w.Q   = af((size_t)AKN * H);
  w.agg = af((size_t)AKN * H);
  w.g   = af((size_t)AKN * H);
  w.S   = af((size_t)AKN * H);
  w.o   = af((size_t)AKN * H);
  w.O   = af((size_t)AKN * H);
  w.H1  = af((size_t)AKN * 512);
  w.Bp  = ah((size_t)128 * 1024);          // 256 KB packed-weight scratch
  _Float16* BpWih = ah((size_t)128 * 384); // prepacked GRU weights
  _Float16* BpWhh = ah((size_t)128 * 384);
  float* Yt = af((size_t)AKN * H);
  float* GI = af((size_t)AKN * 384);
  float* GH = af((size_t)AKN * 384);

  float* out_loc      = (float*)d_out;
  float* out_scale    = out_loc      + (size_t)AKN * FUT_ * OD_;
  float* out_locref   = out_scale    + (size_t)AKN * FUT_ * OD_;
  float* out_scaleref = out_locref   + (size_t)AKN * FUT_ * OD_;
  float* out_pi       = out_scaleref + (size_t)AKN * FUT_ * OD_;

  // ----- edge index arrays (absorb the a2m/m2a major permutations) -----
  build_t2m_edges<<<ew(AKN * NT2M_), 256, 0, st>>>(es_t2m, er_t2m);
  build_deg_edges<<<ew(AKN * DEGPL), 256, 0, st>>>(pl2m_src, DEGPL, es_pl, er_pl);
  build_deg_edges<<<ew(AKN * DEGA), 256, 0, st>>>(a2m_src, DEGA, es_a, er_a);
  build_m2m_edges<<<ew(AKN * KM), 256, 0, st>>>(es_m2m);

  // ----- geometric edge features + fourier embeddings -----
  feat_t2m<<<ew(A_ * NT2M_), 256, 0, st>>>(pos_a, head_a, f_t2m);
  feat_pl2m<<<ew(A_ * DEGPL), 256, 0, st>>>(pos_a, head_a, pos_pl, orient_pl, pl2m_src, f_pl);
  feat_a2m<<<ew(A_ * DEGA), 256, 0, st>>>(pos_a, head_a, a2m_src, f_a);
  fourier_forward(st, Pt2m, f_t2m, 4, A_ * NT2M_, 4, Zb, hb, sb, remb_t2m, w.Bp);
  fourier_forward(st, Ppl,  f_pl,  3, A_ * DEGPL, 3, Zb, hb, sb, remb_pl, w.Bp);
  fourier_forward(st, Pa2m, f_a,   3, A_ * DEGA,  3, Zb, hb, sb, remb_a, w.Bp);

  // ----- proposal stage -----
  init_mode<<<ew(AKN * H), 256, 0, st>>>(mode_emb, mbuf);
  const float* x_a_last = x_a_enc + (size_t)(T_ - 1) * H;  // row stride T_*H
  for (int r = 0; r < RR; ++r) {
    for (int i = 0; i < LL; ++i) {
      attn_apply(st, t2m_prop[i],  x_a_enc, H, A_ * T_, mbuf,
                 remb_t2m, A_ * NT2M_, es_t2m, er_t2m, NT2M_, true, w);
      attn_apply(st, pl2m_prop[i], x_pl, H, P_, mbuf,
                 remb_pl, A_ * DEGPL, es_pl, er_pl, DEGPL, true, w);
      attn_apply(st, a2m_prop[i],  x_a_last, T_ * H, A_, mbuf,
                 remb_a, A_ * DEGA, es_a, er_a, DEGA, true, w);
    }
    attn_apply(st, m2m_prop, mbuf, H, AKN, mbuf, nullptr, 0, es_m2m, nullptr, KM, false, w);
    mlp_head_run(st, to_loc_prop,   mbuf, w.Q, locs   + r * 40, 120, 40, w.Bp);
    mlp_head_run(st, to_scale_prop, mbuf, w.Q, scales + r * 40, 120, 40, w.Bp);
  }
  cumsum_out<<<ew(AKN * OD_), 256, 0, st>>>(locs, scales, out_loc, out_scale);

  // ----- GRU over fourier(y) of the proposed trajectory -----
  bcast_row<<<ew(AKN * H), 256, 0, st>>>(h0, mbuf, AKN * H);
  pack_weights(st, gru.wih, 384, 384, H, BpWih);
  pack_weights(st, gru.whh, 384, 384, H, BpWhh);
  for (int t = 0; t < FUT_; ++t) {
    fourier_forward(st, Py, out_loc + t * OD_, FUT_ * OD_, AKN, OD_, Zb, hb, sb, Yt, w.Bp);
    gemm_packed(st, Yt, H, BpWih, gru.bih, nullptr, 0, nullptr, GI, 384, AKN, 384, H, GF_BIAS);
    gemm_packed(st, mbuf, H, BpWhh, gru.bhh, nullptr, 0, nullptr, GH, 384, AKN, 384, H, GF_BIAS);
    gru_gate<<<ew(AKN * H), 256, 0, st>>>(GI, GH, mbuf, AKN * H);
  }

  // ----- refinement stage -----
  for (int i = 0; i < LL; ++i) {
    attn_apply(st, t2m_ref[i],  x_a_enc, H, A_ * T_, mbuf,
               remb_t2m, A_ * NT2M_, es_t2m, er_t2m, NT2M_, true, w);
    attn_apply(st, pl2m_ref[i], x_pl, H, P_, mbuf,
               remb_pl, A_ * DEGPL, es_pl, er_pl, DEGPL, true, w);
    attn_apply(st, a2m_ref[i],  x_a_last, T_ * H, A_, mbuf,
               remb_a, A_ * DEGA, es_a, er_a, DEGA, true, w);
  }
  attn_apply(st, m2m_ref, mbuf, H, AKN, mbuf, nullptr, 0, es_m2m, nullptr, KM, false, w);

  mlp_head_run(st, to_loc_ref,   mbuf, w.Q, locs,   120, 120, w.Bp);
  mlp_head_run(st, to_scale_ref, mbuf, w.Q, scales, 120, 120, w.Bp);
  ref_out<<<ew(AKN * FUT_ * OD_), 256, 0, st>>>(locs, scales, out_loc,
                                                out_locref, out_scaleref);
  mlp_head_run(st, to_pi, mbuf, w.Q, out_pi, 1, 1, w.Bp);
}
